// GCNModel_70145405878924
// MI455X (gfx1250) — compile-verified, matrix operand order checked by reading, not run on hardware
//
#include <hip/hip_runtime.h>
#include <hip/hip_bf16.h>
#include <stdint.h>

typedef __attribute__((ext_vector_type(16))) _Float16 v16h;
typedef __attribute__((ext_vector_type(8)))  _Float16 v8h;
typedef __attribute__((ext_vector_type(8)))  float    v8f;

union V16H { v16h v; v8h h[2]; };

#define TM 128
#define TN 128
#define TK 32
#define LSTR 48   // LDS row stride in halves (96 bytes: 64B tile row + 32B pad)

#if defined(__gfx1250__) && __has_builtin(__builtin_amdgcn_tensor_load_to_lds) && \
    __has_builtin(__builtin_amdgcn_s_wait_tensorcnt)
#define USE_TDM 1
#else
#define USE_TDM 0
#endif

#if USE_TDM
typedef __attribute__((ext_vector_type(4))) unsigned int u32x4;
typedef __attribute__((ext_vector_type(4))) int i32x4;
typedef __attribute__((ext_vector_type(8))) int i32x8;

// Issue one TDM 2D tile load: 128 rows x 64 bytes (TK halves), row stride K
// halves in memory, 96-byte row stride in LDS via pad_enable.
__device__ __forceinline__ void tdm_load_tile(const _Float16* gsrc, _Float16* ldst, int K)
{
    unsigned long long ga = (unsigned long long)(uintptr_t)gsrc;
    unsigned lds = (unsigned)(uintptr_t)ldst;
    unsigned dim0 = (unsigned)(K >> 2);        // row length in 8-byte units

    u32x4 g0;
    g0[0] = 1u;                                               // count=1, user mode
    g0[1] = lds;                                              // lds_addr
    g0[2] = (unsigned)(ga & 0xffffffffu);                     // global_addr[31:0]
    g0[3] = (unsigned)((ga >> 32) & 0x1ffffffu) | (2u << 30); // [56:32] | type=2

    i32x8 g1;
    g1[0] = (3 << 16) | (1 << 20) | (3 << 22) | (7 << 25);    // 8B elems, pad_en,
                                                              // 16 dw interval, 8 dw pad
    g1[1] = (int)((dim0 & 0xffffu) << 16);                    // tensor_dim0[15:0]
    g1[2] = (int)((dim0 >> 16) | (128u << 16));               // dim0[31:16] | tensor_dim1=128
    g1[3] = (int)(8u << 16);                                  // tile_dim0 = 8 units (64B)
    g1[4] = 128;                                              // tile_dim1 = 128 rows
    g1[5] = (int)dim0;                                        // tensor_dim0_stride[31:0]
    g1[6] = 0;
    g1[7] = 0;

    i32x4 z4 = {0, 0, 0, 0};
#if __clang_major__ >= 23
    i32x8 z8 = {0, 0, 0, 0, 0, 0, 0, 0};
    __builtin_amdgcn_tensor_load_to_lds(g0, g1, z4, z4, z8, 0);
#else
    __builtin_amdgcn_tensor_load_to_lds(g0, g1, z4, z4, 0);
#endif
}
#endif

// fragment loads per ISA 16-bit A 16x32 layout (lane = row, halves 0..7 =
// K[kh..kh+7], halves 8..15 = K[kh+16..kh+23]) + 8 WMMAs
__device__ __forceinline__ void frag_compute(const _Float16* sA, const _Float16* sB,
                                             int wave_m, int wave_n, int lane,
                                             v8f acc[4][2])
{
    const int rl = lane & 15;
    const int kh = (lane >> 4) * 8;
    V16H bfr[2];
    #pragma unroll
    for (int fn = 0; fn < 2; ++fn) {
        const _Float16* p = sB + (wave_n + fn * 16 + rl) * LSTR + kh;
        bfr[fn].h[0] = *(const v8h*)(p);
        bfr[fn].h[1] = *(const v8h*)(p + 16);
    }
    #pragma unroll
    for (int fm = 0; fm < 4; ++fm) {
        V16H afr;
        const _Float16* p = sA + (wave_m + fm * 16 + rl) * LSTR + kh;
        afr.h[0] = *(const v8h*)(p);
        afr.h[1] = *(const v8h*)(p + 16);
        #pragma unroll
        for (int fn = 0; fn < 2; ++fn)
            acc[fm][fn] = __builtin_amdgcn_wmma_f32_16x16x32_f16(
                false, afr.v, false, bfr[fn].v, (short)0, acc[fm][fn], false, false);
    }
}

// ---------------------------------------------------------------------------
// Full-tile NT GEMM (M%128==0, N%128==0, K%32==0): no bounds checks,
// double-buffered LDS, TDM async tile loads when available.
// ---------------------------------------------------------------------------
__global__ __launch_bounds__(256)
void wmma_gemm_full(const _Float16* __restrict__ A, const _Float16* __restrict__ Bw,
                    const float* __restrict__ bias,
                    _Float16* __restrict__ outH, float* __restrict__ outF,
                    int N, int K, int f32mode, int relu)
{
    __shared__ _Float16 sA[2][TM * LSTR];
    __shared__ _Float16 sB[2][TN * LSTR];

    const int tid  = threadIdx.x;
    const int lane = tid & 31;
    const int wid  = tid >> 5;
    const int m0 = blockIdx.y * TM;
    const int n0 = blockIdx.x * TN;
    const int wave_m = (wid >> 2) * 64;
    const int wave_n = (wid & 3) * 32;

    const _Float16* Abase = A  + (size_t)m0 * K;
    const _Float16* Bbase = Bw + (size_t)n0 * K;

    v8f acc[4][2];
    #pragma unroll
    for (int i = 0; i < 4; ++i)
        #pragma unroll
        for (int j = 0; j < 2; ++j) acc[i][j] = {};

    const int nk = K / TK;

#if USE_TDM
    if (wid == 0) {
        tdm_load_tile(Abase, &sA[0][0], K);
        tdm_load_tile(Bbase, &sB[0][0], K);
    }
    for (int it = 0; it < nk; ++it) {
        const int buf = it & 1;
        if (wid == 0) __builtin_amdgcn_s_wait_tensorcnt(0);
        __syncthreads();                       // tile `buf` ready; prior compute
                                               // on buf^1 finished -> safe to DMA
        if (wid == 0 && it + 1 < nk) {
            tdm_load_tile(Abase + (it + 1) * TK, &sA[buf ^ 1][0], K);
            tdm_load_tile(Bbase + (it + 1) * TK, &sB[buf ^ 1][0], K);
        }
        frag_compute(&sA[buf][0], &sB[buf][0], wave_m, wave_n, lane, acc);
    }
#else
    // register-prefetch double buffering (2 b128 loads per thread per tile)
    const int l0 = tid * 2, l1 = tid * 2 + 1;
    const int rA0 = l0 >> 2, gA0 = l0 & 3;
    const int rA1 = l1 >> 2, gA1 = l1 & 3;
    v8h pa0 = *(const v8h*)(Abase + (size_t)rA0 * K + gA0 * 8);
    v8h pa1 = *(const v8h*)(Abase + (size_t)rA1 * K + gA1 * 8);
    v8h pb0 = *(const v8h*)(Bbase + (size_t)rA0 * K + gA0 * 8);
    v8h pb1 = *(const v8h*)(Bbase + (size_t)rA1 * K + gA1 * 8);
    for (int it = 0; it < nk; ++it) {
        const int buf = it & 1;
        *(v8h*)(&sA[buf][0] + rA0 * LSTR + gA0 * 8) = pa0;
        *(v8h*)(&sA[buf][0] + rA1 * LSTR + gA1 * 8) = pa1;
        *(v8h*)(&sB[buf][0] + rA0 * LSTR + gA0 * 8) = pb0;
        *(v8h*)(&sB[buf][0] + rA1 * LSTR + gA1 * 8) = pb1;
        __syncthreads();
        if (it + 1 < nk) {
            const int kt = (it + 1) * TK;
            pa0 = *(const v8h*)(Abase + (size_t)rA0 * K + kt + gA0 * 8);
            pa1 = *(const v8h*)(Abase + (size_t)rA1 * K + kt + gA1 * 8);
            pb0 = *(const v8h*)(Bbase + (size_t)rA0 * K + kt + gA0 * 8);
            pb1 = *(const v8h*)(Bbase + (size_t)rA1 * K + kt + gA1 * 8);
        }
        frag_compute(&sA[buf][0], &sB[buf][0], wave_m, wave_n, lane, acc);
    }
#endif

    // ---- epilogue: no bounds checks on full tiles ----
    const int nl = lane & 15;
    const int mh = (lane >> 4) * 8;
    #pragma unroll
    for (int fm = 0; fm < 4; ++fm) {
        #pragma unroll
        for (int fn = 0; fn < 2; ++fn) {
            const int ng = n0 + wave_n + fn * 16 + nl;
            const float bv = bias ? bias[ng] : 0.f;
            #pragma unroll
            for (int r = 0; r < 8; ++r) {
                const int mg = m0 + wave_m + fm * 16 + mh + r;
                float v = acc[fm][fn][r] + bv;
                if (relu) v = v > 0.f ? v : 0.f;
                const size_t idx = (size_t)mg * N + ng;
                if (outH) outH[idx] = (_Float16)v;
                if (f32mode == 1) outF[idx] = v;
                else if (f32mode == 2) outF[idx] += v;
            }
        }
    }
}

// ---------------------------------------------------------------------------
// Edge GEMM (any M/N): bounds-checked, single buffer. Used only for the
// final 256x1x512 head GEMM.
// ---------------------------------------------------------------------------
__global__ __launch_bounds__(256)
void wmma_gemm_edge(const _Float16* __restrict__ A, const _Float16* __restrict__ Bw,
                    const float* __restrict__ bias,
                    _Float16* __restrict__ outH, float* __restrict__ outF,
                    int M, int N, int K, int f32mode, int relu)
{
    __shared__ _Float16 sA[TM * LSTR];
    __shared__ _Float16 sB[TN * LSTR];

    const int tid  = threadIdx.x;
    const int lane = tid & 31;
    const int wid  = tid >> 5;
    const int m0 = blockIdx.y * TM;
    const int n0 = blockIdx.x * TN;
    const int wave_m = (wid >> 2) * 64;
    const int wave_n = (wid & 3) * 32;

    v8f acc[4][2];
    #pragma unroll
    for (int i = 0; i < 4; ++i)
        #pragma unroll
        for (int j = 0; j < 2; ++j) acc[i][j] = {};

    for (int kt = 0; kt < K; kt += TK) {
        #pragma unroll
        for (int i = 0; i < 2; ++i) {
            int linear = tid * 2 + i;
            int row = linear >> 2;
            int grp = linear & 3;
            v8h za = {};
            v8h zb = {};
            if (m0 + row < M) za = *(const v8h*)(A  + (size_t)(m0 + row) * K + kt + grp * 8);
            if (n0 + row < N) zb = *(const v8h*)(Bw + (size_t)(n0 + row) * K + kt + grp * 8);
            *(v8h*)(sA + row * LSTR + grp * 8) = za;
            *(v8h*)(sB + row * LSTR + grp * 8) = zb;
        }
        __syncthreads();
        frag_compute(sA, sB, wave_m, wave_n, lane, acc);
        __syncthreads();
    }

    const int nl = lane & 15;
    const int mh = (lane >> 4) * 8;
    #pragma unroll
    for (int fm = 0; fm < 4; ++fm) {
        #pragma unroll
        for (int fn = 0; fn < 2; ++fn) {
            int ng = n0 + wave_n + fn * 16 + nl;
            #pragma unroll
            for (int r = 0; r < 8; ++r) {
                int mg = m0 + wave_m + fm * 16 + mh + r;
                if (mg < M && ng < N) {
                    float v = acc[fm][fn][r];
                    if (bias) v += bias[ng];
                    if (relu) v = v > 0.f ? v : 0.f;
                    size_t idx = (size_t)mg * N + ng;
                    if (outH) outH[idx] = (_Float16)v;
                    if (f32mode == 1) outF[idx] = v;
                    else if (f32mode == 2) outF[idx] += v;
                }
            }
        }
    }
}

// ---------------------------------------------------------------------------
// Weight convert+transpose: f32 W[K,N] -> f16 Wt[N,Kpad] (zero-pad K)
// ---------------------------------------------------------------------------
__global__ void k_convert_w(const float* __restrict__ W, _Float16* __restrict__ Wt,
                            int K, int N, int Kpad)
{
    int i = blockIdx.x * blockDim.x + threadIdx.x;
    int total = N * Kpad;
    if (i >= total) return;
    int n = i / Kpad, k = i - n * Kpad;
    Wt[i] = (k < K) ? (_Float16)W[(size_t)k * N + n] : (_Float16)0.0f;
}

// atom_list f32 [65536,39] -> f16 [65536,64] zero-padded
__global__ void k_prep_atom(const float* __restrict__ atom_list, _Float16* __restrict__ X)
{
    int i = blockIdx.x * blockDim.x + threadIdx.x;
    if (i >= 65536 * 64) return;
    int j = i & 63; int a = i >> 6;
    X[i] = (j < 39) ? (_Float16)atom_list[(size_t)a * 39 + j] : (_Float16)0.0f;
}

// gather neighbor (atom||bond) features -> f16 [natoms*6, 64] zero-padded
__global__ void k_gather_nbr(const float* __restrict__ atom_list,
                             const float* __restrict__ bond_list,
                             const int* __restrict__ adeg, const int* __restrict__ bdeg,
                             _Float16* __restrict__ X, int atom0, int natoms)
{
    int i = blockIdx.x * blockDim.x + threadIdx.x;
    int total = natoms * 6 * 64;
    if (i >= total) return;
    int j = i & 63;
    int rd = i >> 6;
    int d = rd % 6;
    int aLoc = rd / 6;
    int a = atom0 + aLoc;
    int b = a >> 8;               // / L  (L = 256)
    float v = 0.f;
    if (j < 39) {
        int ad = adeg[(size_t)a * 6 + d];
        ad = ad < 0 ? 0 : (ad > 255 ? 255 : ad);
        v = atom_list[((size_t)b * 256 + ad) * 39 + j];
    } else if (j < 49) {
        int bd = bdeg[(size_t)a * 6 + d];
        bd = bd < 0 ? 0 : (bd > 255 ? 255 : bd);
        v = bond_list[((size_t)b * 256 + bd) * 10 + (j - 39)];
    }
    X[(size_t)rd * 64 + j] = (_Float16)v;
}

// sum over DEG=6 + relu -> nf f16 [65536,128]
__global__ void k_sumdeg_relu(const float* __restrict__ nfpre, _Float16* __restrict__ nf16,
                              int atom0, int natoms)
{
    int i = blockIdx.x * blockDim.x + threadIdx.x;
    if (i >= natoms * 128) return;
    int c = i & 127; int aLoc = i >> 7;
    float s = 0.f;
    #pragma unroll
    for (int d = 0; d < 6; ++d) s += nfpre[((size_t)aLoc * 6 + d) * 128 + c];
    s = s > 0.f ? s : 0.f;
    nf16[((size_t)(atom0 + aLoc)) * 128 + c] = (_Float16)s;
}

// mol[b,c] = sum_l relu(af[b,l,c]) * mask[b,l]; output f16 for next GEMM
__global__ void k_mol(const float* __restrict__ af, const float* __restrict__ mask,
                      _Float16* __restrict__ molH)
{
    int i = blockIdx.x * blockDim.x + threadIdx.x;
    if (i >= 256 * 128) return;
    int c = i & 127; int b = i >> 7;
    float s = 0.f;
    for (int l = 0; l < 256; ++l) {
        float v = af[((size_t)(b * 256 + l)) * 128 + c];
        v = v > 0.f ? v : 0.f;
        s += v * mask[b * 256 + l];
    }
    molH[(size_t)b * 128 + c] = (_Float16)s;
}

// ---------------------------------------------------------------------------
extern "C" void kernel_launch(void* const* d_in, const int* in_sizes, int n_in,
                              void* d_out, int out_size, void* d_ws, size_t ws_size,
                              hipStream_t stream)
{
    (void)in_sizes; (void)n_in; (void)out_size; (void)ws_size;

    const float* atom_list = (const float*)d_in[0];
    const float* bond_list = (const float*)d_in[1];
    const float* atom_mask = (const float*)d_in[2];
    const int*   adeg      = (const int*)d_in[3];
    const int*   bdeg      = (const int*)d_in[4];
    const float* aw1 = (const float*)d_in[5];  const float* ab1 = (const float*)d_in[6];
    const float* aw2 = (const float*)d_in[7];  const float* ab2 = (const float*)d_in[8];
    const float* nw1 = (const float*)d_in[9];  const float* nb1 = (const float*)d_in[10];
    const float* nw2 = (const float*)d_in[11]; const float* nb2 = (const float*)d_in[12];
    const float* gw1 = (const float*)d_in[13]; const float* gb1 = (const float*)d_in[14];
    const float* gw2 = (const float*)d_in[15]; const float* gb2 = (const float*)d_in[16];
    const float* gw3 = (const float*)d_in[17]; const float* gb3 = (const float*)d_in[18];
    const float* ow1 = (const float*)d_in[19]; const float* ob1 = (const float*)d_in[20];
    const float* ow2 = (const float*)d_in[21]; const float* ob2 = (const float*)d_in[22];
    const float* ow3 = (const float*)d_in[23]; const float* ob3 = (const float*)d_in[24];

    float* afOut   = (float*)d_out;               // [65536,128]
    float* predOut = (float*)d_out + 8388608;     // [256,1]

    size_t off = 0;
    char* base = (char*)d_ws;
    auto alloc = [&](size_t bytes) -> void* {
        void* p = base + off;
        off += (bytes + 255) & ~(size_t)255;
        return p;
    };
    _Float16* aw1t = (_Float16*)alloc(512 * 64 * 2);
    _Float16* aw2t = (_Float16*)alloc(128 * 512 * 2);
    _Float16* nw1t = (_Float16*)alloc(512 * 64 * 2);
    _Float16* nw2t = (_Float16*)alloc(128 * 512 * 2);
    _Float16* gw1t = (_Float16*)alloc((size_t)6 * 512 * 128 * 2);
    _Float16* gw2t = (_Float16*)alloc((size_t)6 * 512 * 512 * 2);
    _Float16* gw3t = (_Float16*)alloc((size_t)6 * 128 * 512 * 2);
    _Float16* ow1t = (_Float16*)alloc(512 * 128 * 2);
    _Float16* ow2t = (_Float16*)alloc(512 * 512 * 2);
    _Float16* ow3t = (_Float16*)alloc(1 * 512 * 2);
    _Float16* xAtom = (_Float16*)alloc((size_t)65536 * 64 * 2);
    _Float16* xNbr  = (_Float16*)alloc((size_t)24576 * 64 * 2);
    _Float16* hbuf  = (_Float16*)alloc((size_t)32768 * 512 * 2);
    _Float16* h2buf = (_Float16*)alloc((size_t)32768 * 512 * 2);
    float*    nfpre = (float*)alloc((size_t)24576 * 128 * 4);
    _Float16* nf16  = (_Float16*)alloc((size_t)65536 * 128 * 2);
    _Float16* molH  = (_Float16*)alloc(256 * 128 * 2);
    _Float16* oh1   = (_Float16*)alloc(256 * 512 * 2);
    _Float16* oh2   = (_Float16*)alloc(256 * 512 * 2);

    auto conv = [&](const float* W, _Float16* Wt, int K, int N, int Kpad) {
        int total = N * Kpad;
        k_convert_w<<<(total + 255) / 256, 256, 0, stream>>>(W, Wt, K, N, Kpad);
    };
    auto gemm = [&](const _Float16* A, const _Float16* Bw, const float* bias,
                    _Float16* outH, float* outF, int M, int N, int K,
                    int f32mode, int relu) {
        dim3 g((N + TN - 1) / TN, (M + TM - 1) / TM);
        if ((M % TM) == 0 && (N % TN) == 0)
            wmma_gemm_full<<<g, 256, 0, stream>>>(A, Bw, bias, outH, outF,
                                                  N, K, f32mode, relu);
        else
            wmma_gemm_edge<<<g, 256, 0, stream>>>(A, Bw, bias, outH, outF,
                                                  M, N, K, f32mode, relu);
    };

    // ---- 1. convert all weights to f16 [N,Kpad] ----
    conv(aw1, aw1t, 39, 512, 64);
    conv(aw2, aw2t, 512, 128, 512);
    conv(nw1, nw1t, 49, 512, 64);
    conv(nw2, nw2t, 512, 128, 512);
    for (int l = 0; l < 6; ++l) {
        conv(gw1 + (size_t)l * 128 * 512, gw1t + (size_t)l * 512 * 128, 128, 512, 128);
        conv(gw2 + (size_t)l * 512 * 512, gw2t + (size_t)l * 512 * 512, 512, 512, 512);
        conv(gw3 + (size_t)l * 512 * 128, gw3t + (size_t)l * 128 * 512, 512, 128, 512);
    }
    conv(ow1, ow1t, 128, 512, 128);
    conv(ow2, ow2t, 512, 512, 512);
    conv(ow3, ow3t, 512, 1, 512);

    // ---- 2. atom_fc: af = relu(X@aw1+ab1)@aw2+ab2 (written into d_out) ----
    k_prep_atom<<<(65536 * 64) / 256, 256, 0, stream>>>(atom_list, xAtom);
    for (int c = 0; c < 2; ++c) {
        const _Float16* Ax = xAtom + (size_t)c * 32768 * 64;
        gemm(Ax, aw1t, ab1, hbuf, nullptr, 32768, 512, 64, 0, 1);
        gemm(hbuf, aw2t, ab2, nullptr, afOut + (size_t)c * 32768 * 128,
             32768, 128, 512, 1, 0);
    }

    // ---- 3. neighbor_fc + sum over DEG + relu -> nf16 ----
    for (int c = 0; c < 16; ++c) {
        int atom0 = c * 4096;
        k_gather_nbr<<<(4096 * 6 * 64) / 256, 256, 0, stream>>>(
            atom_list, bond_list, adeg, bdeg, xNbr, atom0, 4096);
        gemm(xNbr, nw1t, nb1, hbuf, nullptr, 24576, 512, 64, 0, 1);
        gemm(hbuf, nw2t, nb2, nullptr, nfpre, 24576, 128, 512, 1, 0);
        k_sumdeg_relu<<<(4096 * 128) / 256, 256, 0, stream>>>(nfpre, nf16, atom0, 4096);
    }

    // ---- 4. GCN layers: nf = mlp3(nf); af += nf ----
    for (int l = 0; l < 6; ++l) {
        const _Float16* w1 = gw1t + (size_t)l * 512 * 128;
        const _Float16* w2 = gw2t + (size_t)l * 512 * 512;
        const _Float16* w3 = gw3t + (size_t)l * 128 * 512;
        const float* b1 = gb1 + (size_t)l * 512;
        const float* b2 = gb2 + (size_t)l * 512;
        const float* b3 = gb3 + (size_t)l * 128;
        for (int c = 0; c < 2; ++c) {
            _Float16* nfc = nf16 + (size_t)c * 32768 * 128;
            float*    afc = afOut + (size_t)c * 32768 * 128;
            gemm(nfc, w1, b1, hbuf, nullptr, 32768, 512, 128, 0, 1);
            gemm(hbuf, w2, b2, h2buf, nullptr, 32768, 512, 512, 0, 1);
            gemm(h2buf, w3, b3, nfc, afc, 32768, 128, 512, 2, 0);
        }
    }

    // ---- 5. masked pooling + output head ----
    k_mol<<<(256 * 128) / 256, 256, 0, stream>>>(afOut, atom_mask, molH);
    gemm(molH, ow1t, ob1, oh1, nullptr, 256, 512, 128, 0, 1);
    gemm(oh1, ow2t, ob2, oh2, nullptr, 256, 512, 512, 0, 1);
    gemm(oh2, ow3t, ob3, nullptr, predOut, 256, 1, 512, 1, 0);
}
